// MultiheadSelfAttention_28741921144875
// MI455X (gfx1250) — matrix operand order repaired and checked by reading, not compiled, not run on hardware
//
#include <hip/hip_runtime.h>
#include <hip/hip_bf16.h>
#include <math.h>

#define B_   2
#define S_   2048
#define D_   2048
#define H_   16
#define DK_  128
#define BS_  (B_ * S_)

typedef __bf16 bf16;
typedef __attribute__((ext_vector_type(16))) __bf16 v16bf;
typedef __attribute__((ext_vector_type(8)))  float  v8f;
typedef __attribute__((ext_vector_type(4)))  unsigned int u32x4;
typedef __attribute__((ext_vector_type(8)))  unsigned int u32x8;

union FragB16 { v16bf v; uint4 q[2]; };

// A-fragment (16x32 bf16, MxK) from row-major [m][k] source.
__device__ __forceinline__ v16bf load_frag_a(const bf16* base, int ld, int m0, int k0) {
    int lane = threadIdx.x & 31;
    int m = lane & 15, h = lane >> 4;
    const bf16* p = base + (size_t)(m0 + m) * ld + k0;
    FragB16 f;
    f.q[0] = *(const uint4*)(p + 8 * h);
    f.q[1] = *(const uint4*)(p + 16 + 8 * h);
    return f.v;
}

// B-fragment (32x16 bf16, KxN) from row-major [n][k] source (i.e. B^T row-major).
__device__ __forceinline__ v16bf load_frag_b(const bf16* base, int ld, int n0, int k0) {
    int lane = threadIdx.x & 31;
    int n = lane & 15, h = lane >> 4;
    const bf16* p = base + (size_t)(n0 + n) * ld + k0 + 8 * h;
    FragB16 f;
    f.q[0] = *(const uint4*)(p);
    f.q[1] = *(const uint4*)(p + 16);
    return f.v;
}

__device__ __forceinline__ v8f wmma_bf16(v16bf a, v16bf b, v8f c) {
    return __builtin_amdgcn_wmma_f32_16x16x32_bf16(false, a, false, b, (short)0, c, false, false);
}

// Low 32 bits of a generic pointer to __shared__ == LDS byte offset.
__device__ __forceinline__ unsigned lds_off_of(const void* p) {
    return (unsigned)(unsigned long long)p;
}

// ------------------------------------------------- Tensor Data Mover (2D tile)
// Issues TENSOR_LOAD_TO_LDS of a [tile_n=64 rows][tile_k=32 cols] bf16 tile.
// D# per CDNA5 ISA; VADDR2/3 = NULL (2D tensor).
// TDM packs rows contiguously in LDS -> row-major [n][k], ld = 32.
__device__ __forceinline__ void tdm_load_tile64x32(unsigned lds_byte_off,
                                                   const bf16* gtile,
                                                   unsigned dim0_stride_elems) {
    unsigned long long ga = (unsigned long long)gtile;
    u32x4 g0;
    g0[0] = 1u;                                    // count=1, gather off
    g0[1] = lds_byte_off;                          // lds_addr
    g0[2] = (unsigned)ga;                          // global_addr[31:0]
    g0[3] = (unsigned)(ga >> 32) | (2u << 30);     // global_addr[56:32] | type=2
    u32x8 g1;
    g1[0] = 1u << 16;                              // wg_mask=0, data_size=1 (2 bytes)
    g1[1] = (2048u & 0xFFFFu) << 16;               // tensor_dim0[15:0] (k extent)
    g1[2] = (2048u >> 16) | ((2048u & 0xFFFFu) << 16); // td0 hi | tensor_dim1 lo
    g1[3] = (2048u >> 16) | (32u << 16);           // td1 hi | tile_dim0 = 32
    g1[4] = 64u;                                   // tile_dim1 = 64, tile_dim2 = 0
    g1[5] = dim0_stride_elems;                     // tensor_dim0_stride[31:0]
    g1[6] = 0u;                                    // stride0 hi, stride1 lo
    g1[7] = 0u;                                    // stride1 hi
    asm volatile("tensor_load_to_lds %0, %1" :: "s"(g0), "s"(g1) : "memory");
}

// ---------------------------------------------------------------- fp32 -> bf16
__global__ void cvt_bf16(const float* __restrict__ src, bf16* __restrict__ dst, int n) {
    int i = blockIdx.x * blockDim.x + threadIdx.x;
    int stride = gridDim.x * blockDim.x;
    for (; i < n; i += stride) dst[i] = (bf16)src[i];
}

// ------------------------------------------------- cooperative GEMM main loop
// 8 waves (256 thr) compute a 128x64 tile: wave w owns rows row0 = base+16w,
// B tile (64 n x 32 k) is TDM-staged into LDS, double-buffered.
// All 8 ds_load_b128 for the B fragments are issued before the WMMA group so
// the scheduler can clause them and retire with partial s_wait_dscnt.
__device__ __forceinline__ void coop_gemm_128x64(const bf16* __restrict__ X,
                                                 const bf16* __restrict__ W,
                                                 int row0, int o0,
                                                 bf16 (*bt)[64 * 32], v8f acc[4]) {
    bool issuer = (threadIdx.x < 32);  // wave 0 drives the TDM
    unsigned lds0 = lds_off_of(&bt[0][0]);
    unsigned lds1 = lds_off_of(&bt[1][0]);
    const bf16* wbase = W + (size_t)o0 * D_;

    if (issuer) {
        tdm_load_tile64x32(lds0, wbase, D_);
        __builtin_amdgcn_s_wait_tensorcnt(0);
    }
    __syncthreads();

    int cur = 0;
    for (int k0 = 0; k0 < D_; k0 += 32) {
        if ((k0 + 32 < D_) && issuer)
            tdm_load_tile64x32(cur ? lds0 : lds1, wbase + k0 + 32, D_);

        v16bf a = load_frag_a(X, D_, row0, k0);
        const bf16* bsrc = bt[cur];
        v16bf bfr[4];
#pragma unroll
        for (int j = 0; j < 4; ++j) bfr[j] = load_frag_b(bsrc, 32, 16 * j, 0);
#pragma unroll
        for (int j = 0; j < 4; ++j) acc[j] = wmma_bf16(a, bfr[j], acc[j]);

        if (issuer) __builtin_amdgcn_s_wait_tensorcnt(0);
        __syncthreads();   // publish next buffer / protect cur buffer (WAR)
        cur ^= 1;
    }
}

// ------------------------------------------------------- Q/K/V projection GEMM
// grid: (BS/128, D/64, 3), block 256.
__global__ void qkv_proj(const bf16* __restrict__ xb,
                         const bf16* __restrict__ wqb, const bf16* __restrict__ wkb,
                         const bf16* __restrict__ wvb,
                         bf16* __restrict__ Qb, bf16* __restrict__ Kb, bf16* __restrict__ Vt,
                         const int* __restrict__ tpos) {
    __shared__ alignas(128) bf16 bt[2][64 * 32];

    int wave = threadIdx.x >> 5;
    int row0 = blockIdx.x * 128 + wave * 16;   // row in [0, BS)
    int o0   = blockIdx.y * 64;
    int which = blockIdx.z;
    const bf16* w = (which == 0) ? wqb : ((which == 1) ? wkb : wvb);

    v8f acc[4] = {};
    coop_gemm_128x64(xb, w, row0, o0, bt, acc);

    int lane = threadIdx.x & 31;
    int n = lane & 15, h = lane >> 4;
    int bidx = row0 / S_;
    int s_base = row0 % S_;

    if (which == 2) {
#pragma unroll
        for (int j = 0; j < 4; ++j)
#pragma unroll
            for (int r = 0; r < 8; ++r) {
                int s = s_base + r + 8 * h;
                int o = o0 + 16 * j + n;
                int head = o >> 7, d = o & 127;
                Vt[((size_t)(bidx * H_ + head) * DK_ + d) * S_ + s] = (bf16)acc[j][r];
            }
    } else {
        bf16* dst = (which == 0) ? Qb : Kb;
#pragma unroll
        for (int j = 0; j < 4; ++j)
#pragma unroll
            for (int r = 0; r < 8; ++r) {
                float v  = acc[j][r];
                float pv = __shfl_xor(v, 1, 32);  // RoPE partner lives in lane^1
                int s = s_base + r + 8 * h;
                int o = o0 + 16 * j + n;
                int head = o >> 7, d = o & 127;
                float pos  = (float)tpos[s];
                float invf = __expf(-(float)(d & ~1) * (logf(10000.0f) / 128.0f));
                float ang  = pos * invf;
                float sn, cs;
                __sincosf(ang, &sn, &cs);
                float out = (d & 1) ? fmaf(pv, sn, v * cs) : fmaf(v, cs, -pv * sn);
                dst[((size_t)(bidx * H_ + head) * S_ + s) * DK_ + d] = (bf16)out;
            }
    }
}

// ------------------------------------------------------------- flash attention
// grid: (S/16, H, B). One wave per 16-query tile; online softmax over 32-key blocks.
__global__ void flash_attn(const bf16* __restrict__ Qb, const bf16* __restrict__ Kb,
                           const bf16* __restrict__ Vt, bf16* __restrict__ AO) {
    __shared__ alignas(16) bf16 lds_p[16 * 32];

    int q0   = blockIdx.x * 16;
    int head = blockIdx.y;
    int bidx = blockIdx.z;
    const bf16* Qh = Qb + (size_t)(bidx * H_ + head) * S_ * DK_;
    const bf16* Kh = Kb + (size_t)(bidx * H_ + head) * S_ * DK_;
    const bf16* Vh = Vt + (size_t)(bidx * H_ + head) * DK_ * S_;

    int lane = threadIdx.x & 31;
    int n = lane & 15, h = lane >> 4;

    v16bf qa[4];
#pragma unroll
    for (int kc = 0; kc < 4; ++kc) qa[kc] = load_frag_a(Qh, DK_, q0, 32 * kc);

    v8f oacc[8] = {};
    float m_i[8], l_i[8];
#pragma unroll
    for (int r = 0; r < 8; ++r) { m_i[r] = -1e30f; l_i[r] = 0.0f; }

    const float scale = 0.08838834764831845f;  // 1/sqrt(128)

    for (int kb0 = 0; kb0 <= q0 + 15; kb0 += 32) {
        // batch the 8 K-fragment loads ahead of the 8 WMMAs
        v16bf kf[8];
#pragma unroll
        for (int kc = 0; kc < 4; ++kc) {
            kf[2 * kc]     = load_frag_b(Kh, DK_, kb0,      32 * kc);
            kf[2 * kc + 1] = load_frag_b(Kh, DK_, kb0 + 16, 32 * kc);
        }
        v8f sc0 = {}, sc1 = {};
#pragma unroll
        for (int kc = 0; kc < 4; ++kc) {
            sc0 = wmma_bf16(qa[kc], kf[2 * kc],     sc0);
            sc1 = wmma_bf16(qa[kc], kf[2 * kc + 1], sc1);
        }
        float p0[8], p1[8];
#pragma unroll
        for (int r = 0; r < 8; ++r) {
            int q = q0 + r + 8 * h;
            float e0 = ((kb0 + n) <= q)      ? sc0[r] * scale : -1e30f;
            float e1 = ((kb0 + 16 + n) <= q) ? sc1[r] * scale : -1e30f;
            float rm = fmaxf(e0, e1);
#pragma unroll
            for (int mm = 1; mm < 16; mm <<= 1) rm = fmaxf(rm, __shfl_xor(rm, mm, 32));
            float mn = fmaxf(m_i[r], rm);
            float f  = __expf(m_i[r] - mn);
            p0[r] = __expf(e0 - mn);
            p1[r] = __expf(e1 - mn);
            float rs = p0[r] + p1[r];
#pragma unroll
            for (int mm = 1; mm < 16; mm <<= 1) rs += __shfl_xor(rs, mm, 32);
            l_i[r] = l_i[r] * f + rs;
            m_i[r] = mn;
#pragma unroll
            for (int j = 0; j < 8; ++j) oacc[j][r] *= f;
        }
        // Re-layout P (C-fragment) -> A-fragment through LDS
#pragma unroll
        for (int r = 0; r < 8; ++r) {
            lds_p[(r + 8 * h) * 32 + n]      = (bf16)p0[r];
            lds_p[(r + 8 * h) * 32 + 16 + n] = (bf16)p1[r];
        }
        __syncthreads();
        v16bf pa = load_frag_a(lds_p, 32, 0, 0);
        // batch V-fragment loads in groups of 4 ahead of their WMMAs
#pragma unroll
        for (int g = 0; g < 2; ++g) {
            v16bf vf[4];
#pragma unroll
            for (int j = 0; j < 4; ++j)
                vf[j] = load_frag_b(Vh, S_, 16 * (4 * g + j), kb0);
#pragma unroll
            for (int j = 0; j < 4; ++j)
                oacc[4 * g + j] = wmma_bf16(pa, vf[j], oacc[4 * g + j]);
        }
        __syncthreads();
    }

#pragma unroll
    for (int r = 0; r < 8; ++r) {
        float inv = 1.0f / l_i[r];
        int s = q0 + r + 8 * h;
#pragma unroll
        for (int j = 0; j < 8; ++j) {
            int d = 16 * j + n;
            AO[((size_t)bidx * S_ + s) * D_ + head * DK_ + d] = (bf16)(oacc[j][r] * inv);
        }
    }
}

// ------------------------------------------------------------ output projection
// grid: (BS/128, D/64), block 256.
__global__ void out_proj(const bf16* __restrict__ AO, const bf16* __restrict__ wob,
                         float* __restrict__ out) {
    __shared__ alignas(128) bf16 bt[2][64 * 32];

    int wave = threadIdx.x >> 5;
    int row0 = blockIdx.x * 128 + wave * 16;
    int o0   = blockIdx.y * 64;

    v8f acc[4] = {};
    coop_gemm_128x64(AO, wob, row0, o0, bt, acc);

    int lane = threadIdx.x & 31;
    int n = lane & 15, h = lane >> 4;
#pragma unroll
    for (int j = 0; j < 4; ++j)
#pragma unroll
        for (int r = 0; r < 8; ++r)
            out[(size_t)(row0 + r + 8 * h) * D_ + o0 + 16 * j + n] = acc[j][r];
}

extern "C" void kernel_launch(void* const* d_in, const int* in_sizes, int n_in,
                              void* d_out, int out_size, void* d_ws, size_t ws_size,
                              hipStream_t stream) {
    (void)in_sizes; (void)n_in; (void)out_size; (void)ws_size;
    const float* x  = (const float*)d_in[0];
    const float* wq = (const float*)d_in[1];
    const float* wk = (const float*)d_in[2];
    const float* wv = (const float*)d_in[3];
    const float* wo = (const float*)d_in[4];
    const int* tpos = (const int*)d_in[5];
    float* out = (float*)d_out;

    char* ws = (char*)d_ws;
    size_t off = 0;
    bf16* xb  = (bf16*)(ws + off); off += (size_t)BS_ * D_ * 2;   // 16 MB
    bf16* wqb = (bf16*)(ws + off); off += (size_t)D_ * D_ * 2;    // 8 MB
    bf16* wkb = (bf16*)(ws + off); off += (size_t)D_ * D_ * 2;
    bf16* wvb = (bf16*)(ws + off); off += (size_t)D_ * D_ * 2;
    bf16* wob = (bf16*)(ws + off); off += (size_t)D_ * D_ * 2;
    bf16* Qb  = (bf16*)(ws + off); off += (size_t)BS_ * D_ * 2;   // [b][h][s][d]
    bf16* Kb  = (bf16*)(ws + off); off += (size_t)BS_ * D_ * 2;   // [b][h][s][d]
    bf16* Vt  = (bf16*)(ws + off); off += (size_t)BS_ * D_ * 2;   // [b][h][d][s]
    bf16* AO  = (bf16*)(ws + off); off += (size_t)BS_ * D_ * 2;   // [b][s][h*d]

    cvt_bf16<<<dim3(2048), dim3(256), 0, stream>>>(x,  xb,  BS_ * D_);
    cvt_bf16<<<dim3(1024), dim3(256), 0, stream>>>(wq, wqb, D_ * D_);
    cvt_bf16<<<dim3(1024), dim3(256), 0, stream>>>(wk, wkb, D_ * D_);
    cvt_bf16<<<dim3(1024), dim3(256), 0, stream>>>(wv, wvb, D_ * D_);
    cvt_bf16<<<dim3(1024), dim3(256), 0, stream>>>(wo, wob, D_ * D_);

    qkv_proj<<<dim3(BS_ / 128, D_ / 64, 3), dim3(256), 0, stream>>>(
        xb, wqb, wkb, wvb, Qb, Kb, Vt, tpos);

    flash_attn<<<dim3(S_ / 16, H_, B_), dim3(32), 0, stream>>>(Qb, Kb, Vt, AO);

    out_proj<<<dim3(BS_ / 128, D_ / 64), dim3(256), 0, stream>>>(AO, wob, out);
}
